// InterlacedVolume_33328946217107
// MI455X (gfx1250) — matrix operand order repaired and checked by reading, not compiled, not run on hardware
//
#include <hip/hip_runtime.h>

typedef __attribute__((ext_vector_type(16))) __bf16 v16bf;
typedef __attribute__((ext_vector_type(8)))  __bf16 v8bf;
typedef __attribute__((ext_vector_type(8)))  float  v8f;

#define B_  8
#define C_  96
#define H_  64
#define W_  128
#define MAXDISP_ 48

__device__ __forceinline__ v16bf combine16(v8bf lo, v8bf hi) {
  return __builtin_shufflevector(lo, hi, 0,1,2,3,4,5,6,7,8,9,10,11,12,13,14,15);
}
__device__ __forceinline__ v8f wmma_bf16(v16bf a, v16bf b, v8f c) {
  return __builtin_amdgcn_wmma_f32_16x16x32_bf16(false, a, false, b, (short)0, c, false, false);
}

// ---------------------------------------------------------------------------
// Weight pre-swizzle: build lane-major bf16 B-fragments for WMMA 16x16x32.
// B layout (32x16 bf16): lane l holds column n=l&15, K = (l>=16?8:0)+(h&7)+(h>=8?16:0), h=0..15.
// Fragment f occupies halves [f*512, f*512+512): [lane][h].
//   frags 0..2     : L1, kstep = f          (K = tap*8 + c, 72 valid, pad 96)
//   frags 3..74    : L2, f-3 = (tap*4+ks)*2 + ntile   (K = cc = d8*16+oc1, 128)
//   frags 75..101  : L3, f-75 = tap*3 + ks            (K = cc = e*32+oc2, 96)
// ---------------------------------------------------------------------------
__global__ void prep_weights_kernel(const float* __restrict__ w1,
                                    const float* __restrict__ w2,
                                    const float* __restrict__ w3,
                                    __bf16* __restrict__ frags) {
  int t = blockIdx.x * 256 + threadIdx.x;
  if (t >= 102 * 512) return;
  int frag = t >> 9;
  int r    = t & 511;
  int lane = r >> 4;
  int h    = r & 15;
  int n    = lane & 15;
  int k    = ((lane >> 4) << 3) + (h & 7) + ((h >> 3) << 4);   // 0..31 within the 32-K block
  float val = 0.f;
  if (frag < 3) {                       // L1 weights: w1 (16,1,8,3,3)
    int kk = frag * 32 + k;             // 0..95
    if (kk < 72) {
      int tap = kk >> 3, c = kk & 7;
      int ky = tap / 3, kx = tap % 3;
      val = w1[((n * 8 + c) * 3 + ky) * 3 + kx];
    }
  } else if (frag < 75) {               // L2 weights: w2 (32,16,8,3,3)
    int f = frag - 3;
    int ntile = f & 1;
    int tk = f >> 1;
    int tap = tk >> 2, ks = tk & 3;
    int cc = ks * 32 + k;               // 0..127
    int d8 = cc >> 4, oc1 = cc & 15;
    int o = ntile * 16 + n;
    int ky = tap / 3, kx = tap % 3;
    val = w2[(((o * 16 + oc1) * 8 + d8) * 3 + ky) * 3 + kx];
  } else {                              // L3 weights: w3 (16,32,3,3,3)
    int f = frag - 75;
    int tap = f / 3, ks = f % 3;
    int cc = ks * 32 + k;               // 0..95
    int e = cc >> 5, oc2 = cc & 31;
    int ky = tap / 3, kx = tap % 3;
    val = w3[(((n * 32 + oc2) * 3 + e) * 3 + ky) * 3 + kx];
  }
  frags[t] = (__bf16)val;
}

// ---------------------------------------------------------------------------
// Fused cost-volume kernel: one block per (16x16 tile, batch, disparity).
// All intermediate layers computed on a 20x20 grid (25 WMMA M-tiles of 16).
// ---------------------------------------------------------------------------
__global__ __launch_bounds__(256, 1)
void fused_volume_kernel(const float* __restrict__ feat_l,
                         const float* __restrict__ feat_r,
                         const float* __restrict__ s1, const float* __restrict__ b1,
                         const float* __restrict__ s2, const float* __restrict__ b2,
                         const float* __restrict__ s3, const float* __restrict__ b3,
                         const float* __restrict__ w4,
                         const float* __restrict__ s4, const float* __restrict__ b4,
                         const __bf16* __restrict__ frags,
                         float* __restrict__ out) {
  __shared__ __align__(16) __bf16 featbuf[22 * 22 * 8];   //  7.7 KB
  __shared__ __align__(16) __bf16 L1out[400 * 128];       // 102.4 KB
  __shared__ __align__(16) __bf16 L2out[400 * 96];        //  76.8 KB
  __shared__ __align__(16) float  L3out[400 * 16];        //  25.6 KB
  __shared__ float validf[400];                           //   1.6 KB
  __shared__ float P[272];                                // bn params + w4

  const int tile  = blockIdx.x;            // 0..31  (4 x 8 tiles of 16x16)
  const int b     = blockIdx.y;            // 0..7
  const int disp  = blockIdx.z;            // 0..47
  const int tileY = tile >> 3, tileX = tile & 7;
  const int gy0 = tileY * 16 - 2, gx0 = tileX * 16 - 2;   // grid origin
  const int tid  = threadIdx.x;
  const int lane = tid & 31;
  const int wv   = tid >> 5;
  const int hi   = lane >> 4;              // half-wave select
  const int n    = lane & 15;              // output-channel lane / column

  // stage bn params + w4 into LDS
  if (tid < 16) { P[tid] = s1[tid]; P[16 + tid] = b1[tid];
                  P[96 + tid] = s3[tid]; P[112 + tid] = b3[tid]; }
  if (tid < 32) { P[32 + tid] = s2[tid]; P[64 + tid] = b2[tid]; }
  if (tid < 8)  { P[128 + tid] = s4[tid]; P[136 + tid] = b4[tid]; }
  if (tid < 128){ P[144 + tid] = w4[tid]; }
  // precompute validity (image bounds + disparity mask) for the 20x20 grid
  for (int p = tid; p < 400; p += 256) {
    int ry = p / 20, rx = p % 20;
    int grow = gy0 + ry, gcol = gx0 + rx;
    validf[p] = (grow >= 0 && grow < H_ && gcol >= 0 && gcol < W_ && gcol >= disp)
                    ? 1.f : 0.f;
  }

  const float* fL = feat_l + (size_t)b * C_ * H_ * W_;
  const float* fR = feat_r + (size_t)b * C_ * H_ * W_;

  for (int e = 0; e < 3; ++e) {
    for (int d8 = 0; d8 < 8; ++d8) {
      const int d = e * 8 + d8;
      __syncthreads();
      // ---- stage masked interlaced input: featbuf[yy][xx][c], 22x22x8 bf16
      for (int p = tid; p < 22 * 22; p += 256) {
        int yy = p / 22, xx = p % 22;
        int grow = gy0 - 1 + yy, gcol = gx0 - 1 + xx;
        v8bf v = {};
        if (grow >= 0 && grow < H_ && gcol >= 0 && gcol < W_ && gcol >= disp) {
          int base = grow * W_ + gcol;
#pragma unroll
          for (int c = 0; c < 4; ++c) {
            float vl = fL[(size_t)(4 * d + c) * H_ * W_ + base];
            float vr = fR[(size_t)(4 * d + c) * H_ * W_ + base - disp];
            v[2 * c]     = (__bf16)vl;
            v[2 * c + 1] = (__bf16)vr;
          }
        }
        *(v8bf*)&featbuf[p * 8] = v;
      }
      __syncthreads();
      // ---- layer1: 8ch x 9 taps (K=72 pad 96) -> 16ch, on 20x20 grid
      for (int mt = wv; mt < 25; mt += 8) {
        v8f acc = {};
        int m = mt * 16 + n;
        int y = m / 20, x = m % 20;
#pragma unroll
        for (int ks = 0; ks < 3; ++ks) {
          int t0 = ks * 4 + hi;            // tap of low 8-half run
          int t1 = t0 + 2;                 // tap of high 8-half run
          v8bf a0 = {}, a1 = {};
          if (t0 <= 8) a0 = *(const v8bf*)&featbuf[((y + t0 / 3) * 22 + (x + t0 % 3)) * 8];
          if (t1 <= 8) a1 = *(const v8bf*)&featbuf[((y + t1 / 3) * 22 + (x + t1 % 3)) * 8];
          v16bf A = combine16(a0, a1);
          const __bf16* bp = frags + (ks * 512 + lane * 16);
          v16bf Bf = combine16(*(const v8bf*)bp, *(const v8bf*)(bp + 8));
          acc = wmma_bf16(A, Bf, acc);
        }
        float sc = P[n], sh = P[16 + n];
        int mbase = mt * 16 + (hi << 3);
#pragma unroll
        for (int j = 0; j < 8; ++j) {
          int mr = mbase + j;
          float v = fmaxf(fmaf(acc[j], sc, sh), 0.f) * validf[mr];
          L1out[mr * 128 + d8 * 16 + n] = (__bf16)v;
        }
      }
    }
    __syncthreads();
    // ---- layer2: 128ch x 9 taps (K=1152) -> 32ch, for this e
    for (int mt = wv; mt < 25; mt += 8) {
      v8f acc0 = {}, acc1 = {};
      int m = mt * 16 + n;
      int y = m / 20, x = m % 20;
      for (int tap = 0; tap < 9; ++tap) {
        int py = y + tap / 3 - 1, px = x + tap % 3 - 1;
        bool inb = (py >= 0 && py < 20 && px >= 0 && px < 20);
        int base = (py * 20 + px) * 128 + hi * 8;
#pragma unroll
        for (int ks = 0; ks < 4; ++ks) {
          v8bf a0 = {}, a1 = {};
          if (inb) {
            a0 = *(const v8bf*)&L1out[base + ks * 32];
            a1 = *(const v8bf*)&L1out[base + ks * 32 + 16];
          }
          v16bf A = combine16(a0, a1);
          const __bf16* bp = frags + ((size_t)(3 + (tap * 4 + ks) * 2) * 512 + lane * 16);
          v16bf B0 = combine16(*(const v8bf*)bp, *(const v8bf*)(bp + 8));
          v16bf B1 = combine16(*(const v8bf*)(bp + 512), *(const v8bf*)(bp + 520));
          acc0 = wmma_bf16(A, B0, acc0);
          acc1 = wmma_bf16(A, B1, acc1);
        }
      }
      float sc0 = P[32 + n], sh0 = P[64 + n];
      float sc1 = P[48 + n], sh1 = P[80 + n];
      int mbase = mt * 16 + (hi << 3);
#pragma unroll
      for (int j = 0; j < 8; ++j) {
        int mr = mbase + j;
        float vl = validf[mr];
        float v0 = fmaxf(fmaf(acc0[j], sc0, sh0), 0.f) * vl;
        float v1 = fmaxf(fmaf(acc1[j], sc1, sh1), 0.f) * vl;
        L2out[mr * 96 + e * 32 + n]      = (__bf16)v0;
        L2out[mr * 96 + e * 32 + 16 + n] = (__bf16)v1;
      }
    }
    // next e's layer1 rewrites L1out only after the barrier at top of d8 loop
  }
  __syncthreads();
  // ---- layer3: 96ch x 9 taps (K=864) -> 16ch
  for (int mt = wv; mt < 25; mt += 8) {
    v8f acc = {};
    int m = mt * 16 + n;
    int y = m / 20, x = m % 20;
    for (int tap = 0; tap < 9; ++tap) {
      int py = y + tap / 3 - 1, px = x + tap % 3 - 1;
      bool inb = (py >= 0 && py < 20 && px >= 0 && px < 20);
      int base = (py * 20 + px) * 96 + hi * 8;
#pragma unroll
      for (int ks = 0; ks < 3; ++ks) {
        v8bf a0 = {}, a1 = {};
        if (inb) {
          a0 = *(const v8bf*)&L2out[base + ks * 32];
          a1 = *(const v8bf*)&L2out[base + ks * 32 + 16];
        }
        v16bf A = combine16(a0, a1);
        const __bf16* bp = frags + ((size_t)(75 + tap * 3 + ks) * 512 + lane * 16);
        v16bf Bf = combine16(*(const v8bf*)bp, *(const v8bf*)(bp + 8));
        acc = wmma_bf16(A, Bf, acc);
      }
    }
    float sc = P[96 + n], sh = P[112 + n];
    int mbase = mt * 16 + (hi << 3);
#pragma unroll
    for (int j = 0; j < 8; ++j) {
      int mr = mbase + j;
      L3out[mr * 16 + n] = fmaxf(fmaf(acc[j], sc, sh), 0.f) * validf[mr];
    }
  }
  __syncthreads();
  // ---- layer4 (1x1, 16->8) + output store: one thread per output pixel
  {
    int iy = tid >> 4, ix = tid & 15;
    int m = (iy + 2) * 20 + (ix + 2);
    int row = tileY * 16 + iy, col = tileX * 16 + ix;
    float xv[16];
#pragma unroll
    for (int c = 0; c < 16; ++c) xv[c] = L3out[m * 16 + c];
    float mask = (col >= disp) ? 1.f : 0.f;
#pragma unroll
    for (int o = 0; o < 8; ++o) {
      float dot = 0.f;
#pragma unroll
      for (int c = 0; c < 16; ++c) dot += xv[c] * P[144 + o * 16 + c];
      float v = fmaxf(fmaf(dot, P[128 + o], P[136 + o]), 0.f) * mask;
      out[((((size_t)b * 8 + o) * MAXDISP_ + disp) * H_ + row) * W_ + col] = v;
    }
  }
}

extern "C" void kernel_launch(void* const* d_in, const int* in_sizes, int n_in,
                              void* d_out, int out_size, void* d_ws, size_t ws_size,
                              hipStream_t stream) {
  const float* feat_l = (const float*)d_in[0];
  const float* feat_r = (const float*)d_in[1];
  const float* w1 = (const float*)d_in[2];
  const float* s1 = (const float*)d_in[3];
  const float* b1 = (const float*)d_in[4];
  const float* w2 = (const float*)d_in[5];
  const float* s2 = (const float*)d_in[6];
  const float* b2 = (const float*)d_in[7];
  const float* w3 = (const float*)d_in[8];
  const float* s3 = (const float*)d_in[9];
  const float* b3 = (const float*)d_in[10];
  const float* w4 = (const float*)d_in[11];
  const float* s4 = (const float*)d_in[12];
  const float* b4 = (const float*)d_in[13];
  float* out = (float*)d_out;
  __bf16* frags = (__bf16*)d_ws;   // 102 fragments * 1 KB = 104448 bytes

  prep_weights_kernel<<<(102 * 512 + 255) / 256, 256, 0, stream>>>(w1, w2, w3, frags);
  fused_volume_kernel<<<dim3(32, B_, MAXDISP_), 256, 0, stream>>>(
      feat_l, feat_r, s1, b1, s2, b2, s3, b3, w4, s4, b4, frags, out);
}
